// SentBERT_RNN_20066087207122
// MI455X (gfx1250) — matrix operand order repaired and checked by reading, not compiled, not run on hardware
//
#include <hip/hip_runtime.h>
#include <hip/hip_bf16.h>
#include <math.h>
#include <stdint.h>

// ---------------------------------------------------------------------------
// CDNA5 (gfx1250) WMMA helpers: v_wmma_f32_16x16x32_f16, wave32.
// ---------------------------------------------------------------------------
typedef __attribute__((ext_vector_type(16))) _Float16 v16h;
typedef __attribute__((ext_vector_type(8)))  _Float16 v8h;
typedef __attribute__((ext_vector_type(8)))  float    v8f;
typedef __attribute__((ext_vector_type(4)))  unsigned int v4u;
typedef __attribute__((ext_vector_type(8)))  int      v8i;
typedef __attribute__((ext_vector_type(4)))  int      v4i;

__device__ __forceinline__ v8f wmma_f16(v16h a, v16h b, v8f c) {
  // 8 args: (neg_a, A, neg_b, B, c_mod, C, reuse_a, reuse_b)
  return __builtin_amdgcn_wmma_f32_16x16x32_f16(false, a, false, b, (short)0, c,
                                                false, false);
}

// A fragment: 16x32 f16 tile, row-major with leading dim `ld` (elements).
// Lane (0..15):  row = lane,    K = 0..7 and 16..23
// Lane (16..31): row = lane-16, K = 8..15 and 24..31
__device__ __forceinline__ v16h load_a16(const _Float16* tile, int ld) {
  const int lane = threadIdx.x & 31;
  const int row  = lane & 15;
  const int kb   = (lane >> 4) << 3; // 0 or 8
  const _Float16* p = tile + row * ld + kb;
  union { v16h v; v8h h[2]; } u;
  u.h[0] = *(const v8h*)(p);
  u.h[1] = *(const v8h*)(p + 16);
  return u.v;
}

// B fragment: 32x16 f16 tile where B[k][n] = W[n][k], W row-major (N,K), ld=K.
// Lane (0..15):  col = lane,    K = k0+0..7 and k0+16..23
// Lane (16..31): col = lane-16, K = k0+8..15 and k0+24..31
__device__ __forceinline__ v16h load_b16(const _Float16* wrow0, int ld, int k0) {
  const int lane = threadIdx.x & 31;
  const int col  = lane & 15;
  const int kb   = k0 + ((lane >> 4) << 3); // +0 or +8
  const _Float16* p = wrow0 + col * ld + kb;
  union { v16h v; v8h h[2]; } u;
  u.h[0] = *(const v8h*)(p);
  u.h[1] = *(const v8h*)(p + 16);
  return u.v;
}

__device__ __forceinline__ float sigm(float x) {
  return 1.f / (1.f + __expf(-x));
}

// ---------------------------------------------------------------------------
// Problem dims
// ---------------------------------------------------------------------------
#define TT 256
#define BB 64
#define EE 768
#define HH 512
#define H2 1024
#define G4 2048   // 4*H
#define MROWS (TT*BB)  // 16384 flattened (t,b) rows

// ---------------------------------------------------------------------------
// f32 -> f16 conversion
// ---------------------------------------------------------------------------
__global__ void k_cvt(const float* __restrict__ src, _Float16* __restrict__ dst, int n) {
  int i = blockIdx.x * blockDim.x + threadIdx.x;
  if (i < n) dst[i] = (_Float16)src[i];
}

// ---------------------------------------------------------------------------
// K1: pre[m][g] = emb16[m][:] . w16[g][:] + bias[g]
//     M=16384, N=2048, K=768.  One wave computes a 16x64 output strip.
// ---------------------------------------------------------------------------
__global__ void __launch_bounds__(256) k_pre(const _Float16* __restrict__ e16,
                                             const _Float16* __restrict__ w16,
                                             const float* __restrict__ bias,
                                             float* __restrict__ pre) {
  const int JOBS = (MROWS / 16) * (G4 / 64);  // 1024 * 32 = 32768
  const int wid  = (blockIdx.x * blockDim.x + threadIdx.x) >> 5;
  const int nw   = (gridDim.x * blockDim.x) >> 5;
  const int lane = threadIdx.x & 31;
  const int col  = lane & 15;
  const int rb   = (lane >> 4) << 3;

  for (int job = wid; job < JOBS; job += nw) {
    const int mt = job >> 5;
    const int n0 = (job & 31) << 6;
    v8f acc0 = {}, acc1 = {}, acc2 = {}, acc3 = {};
    const _Float16* arow = e16 + (size_t)mt * 16 * EE;
    for (int k = 0; k < EE; k += 32) {
      __builtin_prefetch(arow + k + 128, 0, 0);   // global_prefetch_b8
      v16h a = load_a16(arow + k, EE);
      acc0 = wmma_f16(a, load_b16(w16 + (size_t)(n0 +  0) * EE, EE, k), acc0);
      acc1 = wmma_f16(a, load_b16(w16 + (size_t)(n0 + 16) * EE, EE, k), acc1);
      acc2 = wmma_f16(a, load_b16(w16 + (size_t)(n0 + 32) * EE, EE, k), acc2);
      acc3 = wmma_f16(a, load_b16(w16 + (size_t)(n0 + 48) * EE, EE, k), acc3);
    }
    v8f accs[4] = {acc0, acc1, acc2, acc3};
    float* orow = pre + (size_t)(mt * 16 + rb) * G4 + n0 + col;
    #pragma unroll
    for (int q = 0; q < 4; ++q) {
      const float bv = bias[n0 + q * 16 + col];
      #pragma unroll
      for (int r = 0; r < 8; ++r)
        orow[(size_t)r * G4 + q * 16] = accs[q][r] + bv;
    }
  }
}

// ---------------------------------------------------------------------------
// TDM helper: issue tensor_load_to_lds of a 2D fp16 tile
//   rows x cols (cols contiguous, stride = cols) from `gsrc` into LDS offset 0.
//   D# packing per CDNA5 ISA (group0 / group1).
// ---------------------------------------------------------------------------
#if __has_builtin(__builtin_amdgcn_tensor_load_to_lds)
#define HAVE_TDM 1
__device__ __forceinline__ void tdm_load_f16_2d(const _Float16* gsrc,
                                                unsigned rows, unsigned cols) {
  const unsigned long long ga = (unsigned long long)(uintptr_t)gsrc;
  v4u g0;
  g0[0] = 1u;                                   // count=1 (valid user descriptor)
  g0[1] = 0u;                                   // lds_addr = 0 (dyn-LDS base)
  g0[2] = (unsigned)(ga & 0xFFFFFFFFu);         // global_addr[31:0]
  g0[3] = (unsigned)((ga >> 32) & 0x01FFFFFFu)  // global_addr[56:32]
          | (2u << 30);                         // type = 2 ("image")
  v8i g1;
  g1[0] = (int)(1u << 16);                      // data_size=1 -> 2 bytes
  g1[1] = (int)((cols & 0xFFFFu) << 16);        // tensor_dim0[15:0]
  g1[2] = (int)(((cols >> 16) & 0xFFFFu) | ((rows & 0xFFFFu) << 16)); // dim0 hi | dim1 lo
  g1[3] = (int)(((rows >> 16) & 0xFFFFu) | ((cols & 0xFFFFu) << 16)); // dim1 hi | tile_dim0
  g1[4] = (int)(rows & 0xFFFFu);                // tile_dim1 (tile_dim2 = 0)
  g1[5] = (int)cols;                            // tensor_dim0_stride[31:0]
  g1[6] = 0;                                    // stride hi | dim1_stride lo
  g1[7] = 0;
  v4i z4 = {0, 0, 0, 0};
#if __clang_major__ >= 23
  v8i z8 = {0, 0, 0, 0, 0, 0, 0, 0};
  __builtin_amdgcn_tensor_load_to_lds(g0, g1, z4, z4, z8, 0);
#else
  __builtin_amdgcn_tensor_load_to_lds(g0, g1, z4, z4, 0);
#endif
#if __has_builtin(__builtin_amdgcn_s_wait_tensorcnt)
  __builtin_amdgcn_s_wait_tensorcnt(0);         // s_wait_tensorcnt 0x0
#endif
}
#else
#define HAVE_TDM 0
#endif

// ---------------------------------------------------------------------------
// K2: LSTM recurrence. One workgroup per direction (blockIdx.x = dir).
//     Sequence axis = B (64 steps), batch axis = T (256 rows).
//     h state (256x512) as fp16 in LDS; c state (256x512) fp32 in global ws.
//     Per step: g = pre[:,sb,:] + h @ w_hh^T  (WMMA), gate math in registers,
//     fp16 h staged to global, then TDM (tensor_load_to_lds) refills LDS.
// ---------------------------------------------------------------------------
__global__ void __launch_bounds__(1024) k_rec(const float* __restrict__ pre_f,
                                              const float* __restrict__ pre_b,
                                              const _Float16* __restrict__ whh_f,
                                              const _Float16* __restrict__ whh_b,
                                              float* __restrict__ c_f,
                                              float* __restrict__ c_b,
                                              _Float16* __restrict__ h16s_f,
                                              _Float16* __restrict__ h16s_b,
                                              float* __restrict__ h_all) {
  extern __shared__ _Float16 hs16[];  // 256 x 512 fp16 = 256 KB (WGP LDS = 320 KB)
  const int dir = blockIdx.x;
  const float*     pre   = dir ? pre_b  : pre_f;
  const _Float16*  whh   = dir ? whh_b  : whh_f;
  float*           cst   = dir ? c_b    : c_f;
  _Float16*        h16st = dir ? h16s_b : h16s_f;

  const int tid = threadIdx.x;
  for (int i = tid; i < TT * HH; i += 1024) { hs16[i] = (_Float16)0.f; cst[i] = 0.f; }
  __syncthreads();

  const int wave = tid >> 5;
  const int lane = tid & 31;
  const int col  = lane & 15;
  const int rb   = (lane >> 4) << 3;

  for (int s = 0; s < BB; ++s) {
    const int sb = dir ? (BB - 1 - s) : s;
    // 512 jobs = 16 m-tiles x 32 h-tiles; 16 jobs per wave
    for (int job = wave; job < 512; job += 32) {
      const int mt = job >> 5;
      const int h0 = (job & 31) << 4;
      v8f g[4];
      #pragma unroll
      for (int gi = 0; gi < 4; ++gi)
        #pragma unroll
        for (int r = 0; r < 8; ++r)
          g[gi][r] = pre[((size_t)(mt * 16 + rb + r) * BB + sb) * G4
                         + gi * HH + h0 + col];
      for (int k = 0; k < HH; k += 32) {
        v16h a = load_a16(hs16 + (size_t)mt * 16 * HH + k, HH);
        #pragma unroll
        for (int gi = 0; gi < 4; ++gi)
          g[gi] = wmma_f16(a, load_b16(whh + (size_t)(gi * HH + h0) * HH, HH, k),
                           g[gi]);
      }
      #pragma unroll
      for (int r = 0; r < 8; ++r) {
        const int t = mt * 16 + rb + r;
        const size_t ci = (size_t)t * HH + h0 + col;
        const float ig = sigm(g[0][r]);
        const float fg = sigm(g[1][r]);
        const float gg = tanhf(g[2][r]);
        const float og = sigm(g[3][r]);
        const float c  = fg * cst[ci] + ig * gg;
        cst[ci] = c;
        const float h = og * tanhf(c);
        h_all[((size_t)sb * TT + t) * H2 + dir * HH + h0 + col] = h;
        h16st[ci] = (_Float16)h;     // fp16 staging tile for TDM refill
      }
    }
    __threadfence();
    __syncthreads();
#if HAVE_TDM
    if (tid < 32) tdm_load_f16_2d(h16st, TT, HH);  // DMA 256x512 fp16 -> LDS
#else
    for (int i = tid; i < TT * HH; i += 1024) hs16[i] = h16st[i];
#endif
    __syncthreads();
  }
}

// ---------------------------------------------------------------------------
// K3a: hmean[b][k] = mean_t h_all[b][t][k]
// ---------------------------------------------------------------------------
__global__ void k_mean(const float* __restrict__ h_all, float* __restrict__ hmean) {
  int i = blockIdx.x * blockDim.x + threadIdx.x;
  if (i >= BB * H2) return;
  const int b = i >> 10, k = i & (H2 - 1);
  float s = 0.f;
  for (int t = 0; t < TT; ++t) s += h_all[((size_t)b * TT + t) * H2 + k];
  hmean[i] = s * (1.f / (float)TT);
}

// ---------------------------------------------------------------------------
// K3b: Y[b][j] = opt_tanh( X[b][:] . W[j][:] + bias[j] )   (64x1024 outputs)
// ---------------------------------------------------------------------------
__global__ void k_bwk(const float* __restrict__ X, const float* __restrict__ W,
                      const float* __restrict__ bias, int use_tanh,
                      float* __restrict__ Y) {
  int i = blockIdx.x * blockDim.x + threadIdx.x;
  if (i >= BB * H2) return;
  const int b = i >> 10, j = i & (H2 - 1);
  const float* xv = X + (size_t)b * H2;
  const float* wr = W + (size_t)j * H2;
  float s = bias ? bias[j] : 0.f;
  for (int k = 0; k < H2; ++k) s += xv[k] * wr[k];
  Y[i] = use_tanh ? tanhf(s) : s;
}

// ---------------------------------------------------------------------------
// K3c: s0[t][b] = cont + sal + abs_p[t] + bias
// ---------------------------------------------------------------------------
__global__ void k_scores(const float* __restrict__ h_all,
                         const float* __restrict__ w_content,
                         const float* __restrict__ vsal,
                         const float* __restrict__ pos_emb,
                         const float* __restrict__ w_abs,
                         const float* __restrict__ bias1,
                         float* __restrict__ s0) {
  int i = blockIdx.x * blockDim.x + threadIdx.x;
  if (i >= BB * TT) return;
  const int b = i >> 8, t = i & (TT - 1);
  const float* h = h_all + ((size_t)b * TT + t) * H2;
  float cont = 0.f, sal = 0.f;
  for (int k = 0; k < H2; ++k) {
    cont += h[k] * w_content[k];
    sal  += h[k] * vsal[(size_t)b * H2 + k];
  }
  float ap = 0.f;
  const float* pe = pos_emb + (size_t)t * 100;  // T=256 < PV=1000 -> min() is no-op
  for (int k = 0; k < 100; ++k) ap += pe[k] * w_abs[k];
  s0[t * BB + b] = cont + sal + ap + bias1[0];
}

// ---------------------------------------------------------------------------
// K4: novelty scan. One workgroup, 256 sequential steps.
//     z = tanh(summ) @ W_nov^T via WMMA (u fp16 in LDS),
//     nov[b] = h_t[b] . z[b] via __shfl_xor + ds_add_f32,
//     prob = sigmoid(s0 - nov), summ += prob*h_t.
// ---------------------------------------------------------------------------
__global__ void __launch_bounds__(1024) k_final(const float* __restrict__ h_all,
                                                const _Float16* __restrict__ wnov16,
                                                const float* __restrict__ s0,
                                                float* __restrict__ summ,
                                                float* __restrict__ out) {
  extern __shared__ _Float16 u16[];  // 64 x 1024 fp16 = 128 KB
  __shared__ float nov[BB];
  __shared__ float probs[BB];
  const int tid = threadIdx.x;

  for (int i = tid; i < BB * H2; i += 1024) { summ[i] = 0.f; u16[i] = (_Float16)0.f; }
  if (tid < BB) nov[tid] = 0.f;
  __syncthreads();

  const int wave = tid >> 5;
  const int lane = tid & 31;
  const int col  = lane & 15;
  const int rb   = (lane >> 4) << 3;

  for (int t = 0; t < TT; ++t) {
    // z[b][h] = sum_k u[b][k] * W_nov[h][k]; 256 jobs = 4 m-tiles x 64 n-tiles
    for (int job = wave; job < 256; job += 32) {
      const int mt = job >> 6;
      const int n0 = (job & 63) << 4;
      v8f z = {};
      for (int k = 0; k < H2; k += 32) {
        v16h a = load_a16(u16 + (size_t)mt * 16 * H2 + k, H2);
        z = wmma_f16(a, load_b16(wnov16 + (size_t)n0 * H2, H2, k), z);
      }
      float part[8];
      #pragma unroll
      for (int r = 0; r < 8; ++r) {
        const int b = mt * 16 + rb + r;
        part[r] = z[r] * h_all[((size_t)b * TT + t) * H2 + n0 + col];
      }
      #pragma unroll
      for (int m = 1; m < 16; m <<= 1) {
        #pragma unroll
        for (int r = 0; r < 8; ++r) part[r] += __shfl_xor(part[r], m, 32);
      }
      if (col == 0) {
        #pragma unroll
        for (int r = 0; r < 8; ++r) atomicAdd(&nov[mt * 16 + rb + r], part[r]);
      }
    }
    __syncthreads();
    if (tid < BB) {
      const float p = sigm(s0[t * BB + tid] - nov[tid]);
      probs[tid] = p;
      out[(size_t)tid * TT + t] = p;   // probs.T.reshape(-1)
      nov[tid] = 0.f;                  // reset for next step (next adds post-barrier)
    }
    __syncthreads();
    for (int i = tid; i < BB * H2; i += 1024) {
      const int b = i >> 10;
      const float sv = summ[i] + probs[b] * h_all[((size_t)b * TT + t) * H2 + (i & (H2 - 1))];
      summ[i] = sv;
      u16[i] = (_Float16)tanhf(sv);
    }
    __syncthreads();
  }
}

// ---------------------------------------------------------------------------
// Host launcher
// ---------------------------------------------------------------------------
extern "C" void kernel_launch(void* const* d_in, const int* in_sizes, int n_in,
                              void* d_out, int out_size, void* d_ws, size_t ws_size,
                              hipStream_t stream) {
  (void)in_sizes; (void)n_in; (void)out_size; (void)ws_size;
  const float* emb       = (const float*)d_in[0];
  /* d_in[1] lengths (int) unused */
  const float* w_ih_f    = (const float*)d_in[2];
  const float* w_hh_f    = (const float*)d_in[3];
  const float* b_f       = (const float*)d_in[4];
  const float* w_ih_b    = (const float*)d_in[5];
  const float* w_hh_b    = (const float*)d_in[6];
  const float* b_b       = (const float*)d_in[7];
  const float* W_fdoc    = (const float*)d_in[8];
  const float* b_fdoc    = (const float*)d_in[9];
  const float* pos_emb   = (const float*)d_in[10];
  const float* w_content = (const float*)d_in[11];
  const float* W_sal     = (const float*)d_in[12];
  const float* W_nov     = (const float*)d_in[13];
  const float* w_abs     = (const float*)d_in[14];
  const float* bias1     = (const float*)d_in[15];
  float* out = (float*)d_out;

  char* ws = (char*)d_ws;
  size_t off = 0;
  auto take = [&](size_t bytes) -> char* {
    char* p = ws + off;
    off += (bytes + 255) & ~(size_t)255;
    return p;
  };

  _Float16* e16    = (_Float16*)take((size_t)MROWS * EE * 2);
  _Float16* wihf16 = (_Float16*)take((size_t)G4 * EE * 2);
  _Float16* wihb16 = (_Float16*)take((size_t)G4 * EE * 2);
  _Float16* whhf16 = (_Float16*)take((size_t)G4 * HH * 2);
  _Float16* whhb16 = (_Float16*)take((size_t)G4 * HH * 2);
  _Float16* wnov16 = (_Float16*)take((size_t)H2 * H2 * 2);
  float* pre_f = (float*)take((size_t)MROWS * G4 * 4);
  float* pre_b = (float*)take((size_t)MROWS * G4 * 4);
  float* h_all = (float*)take((size_t)BB * TT * H2 * 4);
  float* c_f   = (float*)take((size_t)TT * HH * 4);
  float* c_b   = (float*)take((size_t)TT * HH * 4);
  _Float16* h16s_f = (_Float16*)take((size_t)TT * HH * 2);
  _Float16* h16s_b = (_Float16*)take((size_t)TT * HH * 2);
  float* hmean = (float*)take((size_t)BB * H2 * 4);
  float* doc   = (float*)take((size_t)BB * H2 * 4);
  float* vsal  = (float*)take((size_t)BB * H2 * 4);
  float* s0    = (float*)take((size_t)TT * BB * 4);
  float* summ  = (float*)take((size_t)BB * H2 * 4);

  // 1) fp32 -> fp16 conversions
  auto cvt = [&](const float* s, _Float16* d, int n) {
    k_cvt<<<(n + 255) / 256, 256, 0, stream>>>(s, d, n);
  };
  cvt(emb,    e16,    MROWS * EE);
  cvt(w_ih_f, wihf16, G4 * EE);
  cvt(w_ih_b, wihb16, G4 * EE);
  cvt(w_hh_f, whhf16, G4 * HH);
  cvt(w_hh_b, whhb16, G4 * HH);
  cvt(W_nov,  wnov16, H2 * H2);

  // 2) input GEMMs (parallel, WMMA)
  k_pre<<<1024, 256, 0, stream>>>(e16, wihf16, b_f, pre_f);
  k_pre<<<1024, 256, 0, stream>>>(e16, wihb16, b_b, pre_b);

  // 3) BiLSTM recurrence: 2 persistent workgroups (fwd, bwd), 256 KB LDS each,
  //    per-step TDM refill of the fp16 h state.
  k_rec<<<2, 1024, TT * HH * sizeof(_Float16), stream>>>(pre_f, pre_b, whhf16,
                                                         whhb16, c_f, c_b,
                                                         h16s_f, h16s_b, h_all);

  // 4) doc vector + saliency
  k_mean<<<(BB * H2 + 255) / 256, 256, 0, stream>>>(h_all, hmean);
  k_bwk<<<(BB * H2 + 255) / 256, 256, 0, stream>>>(hmean, W_fdoc, b_fdoc, 1, doc);
  k_bwk<<<(BB * H2 + 255) / 256, 256, 0, stream>>>(doc, W_sal, nullptr, 0, vsal);
  k_scores<<<(BB * TT + 255) / 256, 256, 0, stream>>>(h_all, w_content, vsal,
                                                      pos_emb, w_abs, bias1, s0);

  // 5) novelty scan (persistent workgroup, WMMA + ds atomics), writes out
  k_final<<<1, 1024, BB * H2 * sizeof(_Float16), stream>>>(h_all, wnov16, s0,
                                                           summ, out);
}
